// PointTransformerLayer_70566312673739
// MI455X (gfx1250) — matrix operand order repaired and checked by reading, not compiled, not run on hardware
//
#include <hip/hip_runtime.h>
#include <hip/hip_bf16.h>

#define B_    4
#define N_    8192
#define DIM_  128
#define K_    16
#define G_    32
#define CPG_  4
#define EPS_  1e-5f
#define CSTR  136   // padded LDS column stride (bf16 elements)

typedef __attribute__((ext_vector_type(16))) __bf16 v16bf;
typedef __attribute__((ext_vector_type(8)))  float  v8f;

// ---------------- CDNA5 async memory->LDS helpers (ASYNCcnt) ----------------
__device__ __forceinline__ void async_ld_b128(void* lds_dst, const float* g) {
  unsigned int loff = (unsigned int)(size_t)lds_dst;   // low 32 bits = LDS offset
  asm volatile("global_load_async_to_lds_b128 %0, %1, off"
               :: "v"(loff), "v"(g) : "memory");
}
__device__ __forceinline__ void wait_async0() {
  asm volatile("s_wait_asynccnt 0" ::: "memory");
}

// ---- WMMA fragment helpers (CDNA5 16x16x32 bf16 layouts, ISA 7.12.2) ----

// A: 16x32 bf16 from f32 source. lane M = lane%16; VGPR i<4: K=2i+off8;
// i>=4: K=16+2(i-4)+off8
__device__ __forceinline__ v16bf cvt_a_frag(const float* __restrict__ W,
                                            int o0, int lane, int cc) {
  int M = lane & 15;
  int off8 = (lane < 16) ? 0 : 8;
  const float* row = W + (o0 + M) * DIM_ + cc;
  v16bf a;
#pragma unroll
  for (int i = 0; i < 8; ++i) {
    int k = (i < 4) ? (2 * i + off8) : (16 + 2 * (i - 4) + off8);
    a[2 * i]     = (__bf16)row[k];
    a[2 * i + 1] = (__bf16)row[k + 1];
  }
  return a;
}

// Pre-shuffled bf16 A-fragment: one 32B v16bf per (matrix m, o-tile, k-step, lane)
__device__ __forceinline__ v16bf load_a_frag_pre(const __bf16* __restrict__ wb,
                                                 int m, int ot, int s, int lane) {
  return *(const v16bf*)(wb + ((((m * 8 + ot) * 4 + s) * 32) + lane) * 16);
}

// B: 32x16 bf16 from column-major LDS (col stride CSTR).
// lane N = lane%16; K = j (+8 for lanes 16..31) for j<8, K = 16+(j-8) (+8) for j>=8
__device__ __forceinline__ v16bf load_b_frag_lds(const __bf16* s, int col0,
                                                 int lane, int cc) {
  int ncol = lane & 15;
  int koff = (lane < 16) ? 0 : 8;
  const __bf16* p = s + (col0 + ncol) * CSTR + cc + koff;
  v16bf b;
#pragma unroll
  for (int j = 0; j < 16; ++j) b[j] = p[j + (j >= 8 ? 8 : 0)];
  return b;
}

// ------ Kernel 0: shuffle Wq/Wk/Wv/Wattn into WMMA-A bf16 fragments ------
__global__ __launch_bounds__(256) void prepw_kernel(
    const float* __restrict__ Wq, const float* __restrict__ Wk,
    const float* __restrict__ Wv, const float* __restrict__ Wattn,
    __bf16* __restrict__ wb) {
  int t = blockIdx.x * 256 + threadIdx.x;   // 4096 threads total
  int lane = t & 31;
  int s = (t >> 5) & 3;
  int ot = (t >> 7) & 7;
  int m = t >> 10;
  const float* Ws[4] = {Wq, Wk, Wv, Wattn};
  v16bf a = cvt_a_frag(Ws[m], ot * 16, lane, 32 * s);
  *(v16bf*)(wb + ((((m * 8 + ot) * 4 + s) * 32) + lane) * 16) = a;
}

// ------- Kernel 1: KNN, double-buffered async LDS tiles (top-16) -------
__global__ __launch_bounds__(256) void knn_kernel(const float* __restrict__ xyz,
                                                  int* __restrict__ idx) {
  const int b = blockIdx.x / (N_ / 256);
  const int n = (blockIdx.x % (N_ / 256)) * 256 + threadIdx.x;
  const float* xb = xyz + b * 3 * N_;
  const float qx = xb[n], qy = xb[N_ + n], qz = xb[2 * N_ + n];

  float dk[K_]; int ik[K_];
#pragma unroll
  for (int j = 0; j < K_; ++j) { dk[j] = 3.4e38f; ik[j] = 0; }

  __shared__ float sb[2][3][1024];
  const int i4 = threadIdx.x * 4;           // 256 threads x 4 floats = 1024
  // prologue: async-load tile 0 into buffer 0
  async_ld_b128(&sb[0][0][i4], xb + i4);
  async_ld_b128(&sb[0][1][i4], xb + N_ + i4);
  async_ld_b128(&sb[0][2][i4], xb + 2 * N_ + i4);
  wait_async0();
  __syncthreads();

  int cur = 0;
  for (int t0 = 0; t0 < N_; t0 += 1024) {
    if (t0 + 1024 < N_) {                   // prefetch next tile asynchronously
      int nxt = cur ^ 1;
      async_ld_b128(&sb[nxt][0][i4], xb + t0 + 1024 + i4);
      async_ld_b128(&sb[nxt][1][i4], xb + N_ + t0 + 1024 + i4);
      async_ld_b128(&sb[nxt][2][i4], xb + 2 * N_ + t0 + 1024 + i4);
    }
    const float* sx = sb[cur][0];
    const float* sy = sb[cur][1];
    const float* sz = sb[cur][2];
    for (int i = 0; i < 1024; ++i) {
      float dx = sx[i] - qx, dy = sy[i] - qy, dz = sz[i] - qz;
      float d = dx * dx + dy * dy + dz * dz;
      if (d < dk[K_ - 1]) {                 // sorted top-16 insertion (static idx)
        int ci = t0 + i; float cd = d;
#pragma unroll
        for (int j = 0; j < K_; ++j) {
          if (cd < dk[j]) {
            float td = dk[j]; int ti = ik[j];
            dk[j] = cd; ik[j] = ci; cd = td; ci = ti;
          }
        }
      }
    }
    wait_async0();
    __syncthreads();
    cur ^= 1;
  }
  int* op = idx + (b * N_ + n) * K_;
#pragma unroll
  for (int j = 0; j < K_; ++j) op[j] = ik[j];
}

// ---- Kernel 2: Q/K/V projection (WMMA bf16), outputs TRANSPOSED (N,C) ----
__global__ __launch_bounds__(256) void qkv_kernel(
    const float* __restrict__ feat, const __bf16* __restrict__ wb,
    const float* __restrict__ bq, const float* __restrict__ bk,
    const float* __restrict__ bv,
    float* __restrict__ q, float* __restrict__ k, float* __restrict__ v) {
  __shared__ __bf16 sF[16 * CSTR];
  const int cg = blockIdx.x * 16;
  const int b = cg / N_;
  const int n0 = cg % N_;
  const float* fb = feat + b * DIM_ * N_ + n0;

#pragma unroll
  for (int i = 0; i < 8; ++i) {                 // stage 128x16 feat tile (bf16)
    int e = i * 256 + threadIdx.x;
    int col = e & 15, c = e >> 4;
    sF[col * CSTR + c] = (__bf16)fb[c * N_ + col];
  }
  __syncthreads();

  const int lane = threadIdx.x & 31, wid = threadIdx.x >> 5;
  const int o0 = wid * 16;
  const int off8 = (lane < 16) ? 0 : 8;
  const int ncol = lane & 15;

  v16bf bf[4];
#pragma unroll
  for (int s = 0; s < 4; ++s) bf[s] = load_b_frag_lds(sF, 0, lane, 32 * s);

  const float* bs[3] = {bq, bk, bv};
  float* outs[3] = {q, k, v};
#pragma unroll
  for (int m = 0; m < 3; ++m) {
    v8f acc = {0.f, 0.f, 0.f, 0.f, 0.f, 0.f, 0.f, 0.f};
#pragma unroll
    for (int s = 0; s < 4; ++s) {
      v16bf a = load_a_frag_pre(wb, m, wid, s, lane);
      acc = __builtin_amdgcn_wmma_f32_16x16x32_bf16(false, a, false, bf[s],
                                                    (short)0, acc, false, false);
    }
    // transposed store: out[(b*N + n), c], lane's 8 rows are contiguous c
    float* op = outs[m] + ((size_t)(b * N_ + n0 + ncol)) * DIM_ + o0 + off8;
#pragma unroll
    for (int r = 0; r < 8; ++r) op[r] = acc[r] + bs[m][o0 + off8 + r];
  }
}

// ------------- Kernel 3: pos_enc GroupNorm statistics (pass 1) -------------
__global__ __launch_bounds__(256) void posstats_kernel(
    const float* __restrict__ xyz, const int* __restrict__ idx,
    const float* __restrict__ Wpos, const float* __restrict__ bpos,
    float* __restrict__ stats) {   // [B][2][32]
  const int b = blockIdx.y;
  __shared__ float sW[384], sB[128], sS0[32], sS1[32];
  for (int i = threadIdx.x; i < 384; i += 256) sW[i] = Wpos[i];
  for (int i = threadIdx.x; i < 128; i += 256) sB[i] = bpos[i];
  if (threadIdx.x < 32) { sS0[threadIdx.x] = 0.f; sS1[threadIdx.x] = 0.f; }
  __syncthreads();

  float acc[32], acc2[32];
#pragma unroll
  for (int g = 0; g < 32; ++g) { acc[g] = 0.f; acc2[g] = 0.f; }

  const float* xb = xyz + b * 3 * N_;
  const int items = N_ * K_;
  for (int it = blockIdx.x * 256 + threadIdx.x; it < items; it += gridDim.x * 256) {
    int n = it >> 4, kk = it & 15;
    int jn = idx[(b * N_ + n) * K_ + kk];
    float dx = xb[jn] - xb[n];
    float dy = xb[N_ + jn] - xb[N_ + n];
    float dz = xb[2 * N_ + jn] - xb[2 * N_ + n];
#pragma unroll
    for (int g = 0; g < 32; ++g) {
      float s0 = 0.f, s1 = 0.f;
#pragma unroll
      for (int cc = 0; cc < 4; ++cc) {
        int c = g * 4 + cc;
        float u = sW[c * 3] * dx + sW[c * 3 + 1] * dy + sW[c * 3 + 2] * dz + sB[c];
        s0 += u; s1 += u * u;
      }
      acc[g] += s0; acc2[g] += s1;
    }
  }
#pragma unroll
  for (int g = 0; g < 32; ++g) {
    atomicAdd(&sS0[g], acc[g]);
    atomicAdd(&sS1[g], acc2[g]);
  }
  __syncthreads();
  if (threadIdx.x < 32) {
    atomicAdd(&stats[b * 64 + threadIdx.x], sS0[threadIdx.x]);
    atomicAdd(&stats[b * 64 + 32 + threadIdx.x], sS1[threadIdx.x]);
  }
}

// ---- helper: build S = k_j - q + pos_enc for 4 points (64 columns) ----
// q/kk are TRANSPOSED (N,C): contiguous 128B vector loads per column slice
__device__ __forceinline__ void build_s_tile(
    int b, int n0, int tid, const float* xyz, const int* idx,
    const float* q, const float* kk,
    const float* sW, const float* sBp, const float* sGp, const float* sBpg,
    const float* sMuP, const float* sRsP,
    __bf16* sS, __bf16* sP /*nullable*/, int* sIdx /*nullable*/) {
  const int col = tid >> 2;      // 0..63
  const int sub = tid & 3;       // 0..3 -> 32 channels each
  const int pi = col >> 4, kki = col & 15;
  const int n = n0 + pi;
  const int jn = idx[(b * N_ + n) * K_ + kki];
  if (sIdx && sub == 0) sIdx[col] = jn;
  const float* xb = xyz + b * 3 * N_;
  float dx = xb[jn] - xb[n];
  float dy = xb[N_ + jn] - xb[N_ + n];
  float dz = xb[2 * N_ + jn] - xb[2 * N_ + n];
  const float4* kj = (const float4*)(kk + ((size_t)(b * N_ + jn)) * DIM_ + sub * 32);
  const float4* qn = (const float4*)(q  + ((size_t)(b * N_ + n )) * DIM_ + sub * 32);
#pragma unroll
  for (int i8 = 0; i8 < 8; ++i8) {
    float4 kv = kj[i8], qv = qn[i8];
    int g = sub * 8 + i8;
    float mu = sMuP[g], rs = sRsP[g];
    int c0 = sub * 32 + i8 * 4;
    float kvv[4] = {kv.x, kv.y, kv.z, kv.w};
    float qvv[4] = {qv.x, qv.y, qv.z, qv.w};
#pragma unroll
    for (int cc = 0; cc < 4; ++cc) {
      int c = c0 + cc;
      float u = sW[c * 3] * dx + sW[c * 3 + 1] * dy + sW[c * 3 + 2] * dz + sBp[c];
      float p = fmaxf((u - mu) * rs * sGp[c] + sBpg[c], 0.f);
      float s = kvv[cc] - qvv[cc] + p;
      sS[col * CSTR + c] = (__bf16)s;
      if (sP) sP[col * CSTR + c] = (__bf16)p;
    }
  }
}

// ------------- Kernel 4: attn GroupNorm statistics (WMMA pass) -------------
__global__ __launch_bounds__(256) void attnstats_kernel(
    const float* __restrict__ xyz, const int* __restrict__ idx,
    const float* __restrict__ q, const float* __restrict__ kk,
    const __bf16* __restrict__ wb,
    const float* __restrict__ Wpos, const float* __restrict__ bpos,
    const float* __restrict__ gpos, const float* __restrict__ bposgn,
    const float* __restrict__ battn,
    const float* __restrict__ posstats, float* __restrict__ attnstats) {
  __shared__ __bf16 sS[64 * CSTR];
  __shared__ float sW[384], sBp[128], sGp[128], sBpg[128], sBat[128];
  __shared__ float sMuP[32], sRsP[32], sA0[32], sA1[32];
  const int tid = threadIdx.x;
  const int cg = blockIdx.x * 4;
  const int b = cg / N_;
  const int n0 = cg % N_;

  for (int i = tid; i < 384; i += 256) sW[i] = Wpos[i];
  for (int i = tid; i < 128; i += 256) {
    sBp[i] = bpos[i]; sGp[i] = gpos[i]; sBpg[i] = bposgn[i]; sBat[i] = battn[i];
  }
  if (tid < 32) {
    const float cnt = (float)(CPG_ * N_ * K_);
    float mu = posstats[b * 64 + tid] / cnt;
    float var = posstats[b * 64 + 32 + tid] / cnt - mu * mu;
    sMuP[tid] = mu; sRsP[tid] = rsqrtf(var + EPS_);
    sA0[tid] = 0.f; sA1[tid] = 0.f;
  }
  __syncthreads();
  build_s_tile(b, n0, tid, xyz, idx, q, kk, sW, sBp, sGp, sBpg, sMuP, sRsP,
               sS, nullptr, nullptr);
  __syncthreads();

  const int lane = tid & 31, wid = tid >> 5;
  const int o0 = wid * 16;
  const int off8 = (lane < 16) ? 0 : 8;
  const int goff = (lane < 16) ? 0 : 2;
  v16bf aw[4];
#pragma unroll
  for (int s = 0; s < 4; ++s) aw[s] = load_a_frag_pre(wb, 3, wid, s, lane);

  float sLo = 0.f, sLo2 = 0.f, sHi = 0.f, sHi2 = 0.f;  // rows r0-3 / r4-7
  for (int ct = 0; ct < 4; ++ct) {
    v8f acc = {0.f, 0.f, 0.f, 0.f, 0.f, 0.f, 0.f, 0.f};
#pragma unroll
    for (int s = 0; s < 4; ++s) {
      v16bf bfr = load_b_frag_lds(sS, ct * 16, lane, 32 * s);
      acc = __builtin_amdgcn_wmma_f32_16x16x32_bf16(false, aw[s], false, bfr,
                                                    (short)0, acc, false, false);
    }
#pragma unroll
    for (int r = 0; r < 8; ++r) {
      float t = acc[r] + sBat[o0 + r + off8];
      if (r < 4) { sLo += t; sLo2 += t * t; } else { sHi += t; sHi2 += t * t; }
    }
  }
  atomicAdd(&sA0[(o0 >> 2) + goff + 0], sLo);
  atomicAdd(&sA1[(o0 >> 2) + goff + 0], sLo2);
  atomicAdd(&sA0[(o0 >> 2) + goff + 1], sHi);
  atomicAdd(&sA1[(o0 >> 2) + goff + 1], sHi2);
  __syncthreads();
  if (tid < 32) {
    atomicAdd(&attnstats[b * 64 + tid], sA0[tid]);
    atomicAdd(&attnstats[b * 64 + 32 + tid], sA1[tid]);
  }
}

// ------- Kernel 5: recompute + softmax + aggregate + y + final stats -------
__global__ __launch_bounds__(256) void final_kernel(
    const float* __restrict__ xyz, const int* __restrict__ idx,
    const float* __restrict__ q, const float* __restrict__ kk,
    const float* __restrict__ v, const float* __restrict__ feat,
    const __bf16* __restrict__ wb,
    const float* __restrict__ Wpos, const float* __restrict__ bpos,
    const float* __restrict__ gpos, const float* __restrict__ bposgn,
    const float* __restrict__ battn,
    const float* __restrict__ gattn, const float* __restrict__ battngn,
    const float* __restrict__ posstats, const float* __restrict__ attnstats,
    float* __restrict__ y, float* __restrict__ finstats) {
  __shared__ __bf16 sS[64 * CSTR];
  __shared__ __bf16 sP[64 * CSTR];
  __shared__ float sW[384], sBp[128], sGp[128], sBpg[128], sBat[128];
  __shared__ float sGa[128], sBag[128];
  __shared__ float sMuP[32], sRsP[32], sMuA[32], sRsA[32], sF0[32], sF1[32];
  __shared__ float sLog[64], sAtt[64];
  __shared__ int sIdx[64];
  const int tid = threadIdx.x;
  const int cg = blockIdx.x * 4;
  const int b = cg / N_;
  const int n0 = cg % N_;

  for (int i = tid; i < 384; i += 256) sW[i] = Wpos[i];
  for (int i = tid; i < 128; i += 256) {
    sBp[i] = bpos[i]; sGp[i] = gpos[i]; sBpg[i] = bposgn[i];
    sBat[i] = battn[i]; sGa[i] = gattn[i]; sBag[i] = battngn[i];
  }
  if (tid < 32) {
    const float cnt = (float)(CPG_ * N_ * K_);
    float mu = posstats[b * 64 + tid] / cnt;
    float var = posstats[b * 64 + 32 + tid] / cnt - mu * mu;
    sMuP[tid] = mu; sRsP[tid] = rsqrtf(var + EPS_);
    float mua = attnstats[b * 64 + tid] / cnt;
    float vara = attnstats[b * 64 + 32 + tid] / cnt - mua * mua;
    sMuA[tid] = mua; sRsA[tid] = rsqrtf(vara + EPS_);
    sF0[tid] = 0.f; sF1[tid] = 0.f;
  }
  if (tid < 64) sLog[tid] = 0.f;
  __syncthreads();
  build_s_tile(b, n0, tid, xyz, idx, q, kk, sW, sBp, sGp, sBpg, sMuP, sRsP,
               sS, sP, sIdx);
  __syncthreads();

  const int lane = tid & 31, wid = tid >> 5;
  const int o0 = wid * 16;
  const int off8 = (lane < 16) ? 0 : 8;
  const int ncol = lane & 15;
  v16bf aw[4];
#pragma unroll
  for (int s = 0; s < 4; ++s) aw[s] = load_a_frag_pre(wb, 3, wid, s, lane);

  for (int ct = 0; ct < 4; ++ct) {
    v8f acc = {0.f, 0.f, 0.f, 0.f, 0.f, 0.f, 0.f, 0.f};
#pragma unroll
    for (int s = 0; s < 4; ++s) {
      v16bf bfr = load_b_frag_lds(sS, ct * 16, lane, 32 * s);
      acc = __builtin_amdgcn_wmma_f32_16x16x32_bf16(false, aw[s], false, bfr,
                                                    (short)0, acc, false, false);
    }
    float hsum = 0.f;
#pragma unroll
    for (int r = 0; r < 8; ++r) {
      int o = o0 + r + off8;
      float t = acc[r] + sBat[o];
      int g = o >> 2;
      float h = (t - sMuA[g]) * sRsA[g] * sGa[o] + sBag[o];
      hsum += fmaxf(h, 0.f);
    }
    atomicAdd(&sLog[ct * 16 + ncol], hsum);
  }
  __syncthreads();

  if (tid < 4) {                          // softmax over K per point
    float m = -3.4e38f;
#pragma unroll
    for (int j = 0; j < 16; ++j) m = fmaxf(m, sLog[tid * 16 + j]);
    float e[16]; float sum = 0.f;
#pragma unroll
    for (int j = 0; j < 16; ++j) { e[j] = __expf(sLog[tid * 16 + j] - m); sum += e[j]; }
    float inv = 1.f / sum;
#pragma unroll
    for (int j = 0; j < 16; ++j) sAtt[tid * 16 + j] = e[j] * inv;
  }
  __syncthreads();

  const float* vb = v + ((size_t)b) * N_ * DIM_;   // transposed (N,C)
  const float* fb = feat + b * DIM_ * N_;
  float* yb = y + b * DIM_ * N_;
#pragma unroll
  for (int rep = 0; rep < 2; ++rep) {     // 512 (c,pi) items over 256 threads
    int item = rep * 256 + tid;
    int c = item >> 2, pi = item & 3;
    int n = n0 + pi;
    float a = 0.f;
#pragma unroll
    for (int j = 0; j < 16; ++j) {
      int col = pi * 16 + j;
      int jn = sIdx[col];
      float p = (float)sP[col * CSTR + c];
      a += (vb[(size_t)jn * DIM_ + c] + p) * sAtt[col];
    }
    float yv = fb[c * N_ + n] + a;
    yb[c * N_ + n] = yv;
    atomicAdd(&sF0[c >> 2], yv);
    atomicAdd(&sF1[c >> 2], yv * yv);
  }
  __syncthreads();
  if (tid < 32) {
    atomicAdd(&finstats[b * 64 + tid], sF0[tid]);
    atomicAdd(&finstats[b * 64 + 32 + tid], sF1[tid]);
  }
}

// ---------------- Kernel 6: final GroupNorm + ReLU ----------------
__global__ __launch_bounds__(256) void finalize_kernel(
    const float* __restrict__ y, const float* __restrict__ finstats,
    const float* __restrict__ gnorm, const float* __restrict__ bnorm,
    float* __restrict__ out) {
  int i = blockIdx.x * 256 + threadIdx.x;
  int c = (i / N_) % DIM_;
  int b = i / (DIM_ * N_);
  int g = c >> 2;
  const float cnt = (float)(CPG_ * N_);
  float mu = finstats[b * 64 + g] / cnt;
  float var = finstats[b * 64 + 32 + g] / cnt - mu * mu;
  float rs = rsqrtf(var + EPS_);
  float val = (y[i] - mu) * rs * gnorm[c] + bnorm[c];
  out[i] = fmaxf(val, 0.f);
}

extern "C" void kernel_launch(void* const* d_in, const int* in_sizes, int n_in,
                              void* d_out, int out_size, void* d_ws, size_t ws_size,
                              hipStream_t stream) {
  (void)in_sizes; (void)n_in; (void)out_size; (void)ws_size;
  const float* xyz     = (const float*)d_in[0];
  const float* feat    = (const float*)d_in[1];
  const float* Wq      = (const float*)d_in[2];
  const float* bq      = (const float*)d_in[3];
  const float* Wk      = (const float*)d_in[4];
  const float* bk      = (const float*)d_in[5];
  const float* Wv      = (const float*)d_in[6];
  const float* bv      = (const float*)d_in[7];
  const float* Wpos    = (const float*)d_in[8];
  const float* bpos    = (const float*)d_in[9];
  const float* gpos    = (const float*)d_in[10];
  const float* bposgn  = (const float*)d_in[11];
  const float* Wattn   = (const float*)d_in[12];
  const float* battn   = (const float*)d_in[13];
  const float* gattn   = (const float*)d_in[14];
  const float* battngn = (const float*)d_in[15];
  const float* gnorm   = (const float*)d_in[16];
  const float* bnorm   = (const float*)d_in[17];

  char* ws = (char*)d_ws;
  int*    d_idx = (int*)ws;                                  // 2 MB
  float*  d_q   = (float*)(ws + (size_t)(2)  * (1u << 20));  // 16 MB (N,C)
  float*  d_k   = (float*)(ws + (size_t)(18) * (1u << 20));  // 16 MB (N,C)
  float*  d_v   = (float*)(ws + (size_t)(34) * (1u << 20));  // 16 MB (N,C)
  float*  d_y   = (float*)(ws + (size_t)(50) * (1u << 20));  // 16 MB (C,N)
  float*  d_pst = (float*)(ws + (size_t)(66) * (1u << 20));  // 3x256 f stats
  float*  d_ast = d_pst + 256;
  float*  d_fst = d_pst + 512;
  __bf16* d_wb  = (__bf16*)(ws + (size_t)(66) * (1u << 20) + 65536); // 128 KB

  hipMemsetAsync(d_pst, 0, 768 * sizeof(float), stream);

  prepw_kernel<<<16, 256, 0, stream>>>(Wq, Wk, Wv, Wattn, d_wb);
  knn_kernel<<<B_ * (N_ / 256), 256, 0, stream>>>(xyz, d_idx);
  qkv_kernel<<<B_ * N_ / 16, 256, 0, stream>>>(feat, d_wb, bq, bk, bv,
                                               d_q, d_k, d_v);
  posstats_kernel<<<dim3(64, B_), 256, 0, stream>>>(xyz, d_idx, Wpos, bpos, d_pst);
  attnstats_kernel<<<B_ * N_ / 4, 256, 0, stream>>>(
      xyz, d_idx, d_q, d_k, d_wb, Wpos, bpos, gpos, bposgn, battn, d_pst, d_ast);
  final_kernel<<<B_ * N_ / 4, 256, 0, stream>>>(
      xyz, d_idx, d_q, d_k, d_v, feat, d_wb, Wpos, bpos, gpos, bposgn,
      battn, gattn, battngn, d_pst, d_ast, d_y, d_fst);
  finalize_kernel<<<B_ * DIM_ * N_ / 256, 256, 0, stream>>>(
      d_y, d_fst, gnorm, bnorm, (float*)d_out);
}
